// Encoder_6777458393829
// MI455X (gfx1250) — matrix operand order repaired and checked by reading, not compiled, not run on hardware
//
#include <hip/hip_runtime.h>
#include <hip/hip_bf16.h>

typedef __attribute__((ext_vector_type(16))) _Float16 v16h;
typedef __attribute__((ext_vector_type(8)))  float    v8f;

#define NN 50000
#define NE 800000

// d_out float offsets: (h0[NN,64], v0[NN,64,3], ea[NE,64]) flattened
#define OUT_H0 0
#define OUT_V0 (NN * 64)
#define OUT_EA (NN * 64 + NN * 64 * 3)

// workspace byte offsets
#define WS_FRAGS 0
#define WS_F16F  131072
#define WS_AGGS  (131072 + NN * 64 * 2)
#define WS_EA16  (WS_AGGS + NN * 64 * 4)
#define WS_NEED_EA16 ((size_t)WS_EA16 + (size_t)NE * 64 * 2)

// B-fragment base indices (1 frag = 32 lanes x 16 halves = 1024 B)
#define FRAG_WB1 0
#define FRAG_WB2 4
#define FRAG_WS1 12
#define FRAG_WS2 36
#define FRAG_WH1 44
#define FRAG_WH2 60
#define FRAG_WV1 68
#define FRAG_WV2 92
#define N_FRAGS  100

struct __align__(32) AF {
  union { v16h v; _Float16 h[16]; uint4 q[2]; };
};

// Load this lane's pre-packed B fragment (2x global_load_b128, L2 resident).
__device__ __forceinline__ v16h bfrag(const _Float16* frags, int fi, int lane) {
  return *(const v16h*)(frags + (size_t)fi * 512 + lane * 16);
}

// A fragment (16x32 f16, ISA layout): lane (g = lane/16, M = lane%16),
// element e -> K = colBase + (e<8 ? 8g+e : 8 + 8g + e). Two contiguous
// 16B pieces per lane.
__device__ __forceinline__ void a_f16(AF& a, const _Float16* row, int colBase, int g) {
  a.q[0] = *(const uint4*)(row + colBase + 8 * g);
  a.q[1] = *(const uint4*)(row + colBase + 16 + 8 * g);
}
__device__ __forceinline__ void a_f32(AF& a, const float* row, int colBase, int g) {
  const float* p0 = row + colBase + 8 * g;
  const float* p1 = row + colBase + 16 + 8 * g;
#pragma unroll
  for (int k = 0; k < 8; ++k) { a.h[k] = (_Float16)p0[k]; a.h[8 + k] = (_Float16)p1[k]; }
}

__device__ __forceinline__ v8f wmma16(const AF& a, v16h b, v8f c) {
  return __builtin_amdgcn_wmma_f32_16x16x32_f16(false, a.v, false, b, (short)0, c, false, false);
}

// Fast SiLU: x * rcp(1+exp(-x)) -> v_mul, v_exp_f32, v_rcp_f32, v_mul.
// (Avoids the IEEE-correct v_div_scale/v_div_fmas/v_div_fixup sequence.)
__device__ __forceinline__ float silu(float x) {
  return x * __builtin_amdgcn_rcpf(1.0f + __expf(-x));
}

#define WAIT_DS() asm volatile("s_wait_dscnt 0x0" ::: "memory")

// ---------------------------------------------------------------------------
// Repack weight matrices [K,64] f32 -> f16 WMMA B fragments in workspace.
// B layout (K x 16 tile): lane (g, n), element e -> K = 32*kc + 8*g + (e<8 ? e : 8+e).
// ---------------------------------------------------------------------------
__global__ void __launch_bounds__(32) k_prep(
    const float* Wb1, const float* Wb2, const float* Ws1, const float* Ws2,
    const float* Wh1, const float* Wh2, const float* Wv1, const float* Wv2,
    _Float16* frags) {
  int b = blockIdx.x;           // fragment id, 0..99
  int lane = threadIdx.x;       // 0..31
  const int cum[9] = {0, 4, 12, 36, 44, 60, 68, 92, 100};
  const int kr[8] = {16, 64, 192, 64, 128, 64, 192, 64};
  const float* srcs[8] = {Wb1, Wb2, Ws1, Ws2, Wh1, Wh2, Wv1, Wv2};
  int m = 0;
#pragma unroll
  for (int q = 1; q < 8; ++q) if (b >= cum[q]) m = q;
  int local = b - cum[m];
  int kc = local >> 2, t = local & 3;
  int g = lane >> 4, n = lane & 15;
  const float* W = srcs[m];
  int K0 = 32 * kc + 8 * g;
  AF h;
#pragma unroll
  for (int e = 0; e < 16; ++e) {
    int K = K0 + (e < 8 ? e : 8 + e);
    h.h[e] = (K < kr[m]) ? (_Float16)W[(size_t)K * 64 + 16 * t + n] : (_Float16)0.0f;
  }
  _Float16* dst = frags + (size_t)b * 512 + lane * 16;
  *(uint4*)(dst) = h.q[0];
  *(uint4*)(dst + 8) = h.q[1];
}

// ---------------------------------------------------------------------------
// Zero the atomic accumulators: agg_s (ws) and v0 (output region).
// ---------------------------------------------------------------------------
__global__ void __launch_bounds__(256) k_zero(float* aggs, float* v0) {
  size_t stride = (size_t)gridDim.x * blockDim.x;
  const size_t nAgg = (size_t)NN * 64;
  const size_t total = nAgg + (size_t)NN * 192;
  for (size_t i = (size_t)blockIdx.x * blockDim.x + threadIdx.x; i < total; i += stride) {
    if (i < nAgg) aggs[i] = 0.0f;
    else v0[i - nAgg] = 0.0f;
  }
}

// ---------------------------------------------------------------------------
// f = species @ Wa + ba  (K=5, tiny) -> f16 table for WMMA gathers.
// ---------------------------------------------------------------------------
__global__ void __launch_bounds__(256) k_node_embed(
    const float* species, const float* Wa, const float* ba, _Float16* f16f) {
  int idx = blockIdx.x * blockDim.x + threadIdx.x;
  if (idx >= NN * 64) return;
  int node = idx >> 6, d = idx & 63;
  const float* s = species + (size_t)node * 5;
  float acc = ba[d];
#pragma unroll
  for (int k = 0; k < 5; ++k) acc += s[k] * Wa[k * 64 + d];
  f16f[idx] = (_Float16)acc;
}

// ---------------------------------------------------------------------------
// ea = MLP(edge_attr; Wb1,bb1,Wb2,bb2).  16 edges / wave, 8 waves / block.
// Optionally also writes an f16 copy of ea for the message kernel.
// ---------------------------------------------------------------------------
__global__ void __launch_bounds__(256) k_edge_embed(
    const float* edge_attr, const float* bb1, const float* bb2,
    const _Float16* frags, float* ea_out, _Float16* ea16) {
  int wave = (blockIdx.x * blockDim.x + threadIdx.x) >> 5;
  int lane = threadIdx.x & 31;
  int w = threadIdx.x >> 5;
  int base = wave * 16;                 // NE = 50000*16, exact
  int g = lane >> 4, n = lane & 15;

  // Layer 1: K=16 zero-padded to 32.
  AF a;
  {
    const float* row = edge_attr + (size_t)(base + n) * 16;
    const float* p = row + 8 * g;
#pragma unroll
    for (int k = 0; k < 8; ++k) { a.h[k] = (_Float16)p[k]; a.h[8 + k] = (_Float16)0.0f; }
  }
  __shared__ __align__(16) _Float16 hm[8][16][64];
#pragma unroll
  for (int t = 0; t < 4; ++t) {
    v8f c = {};
    c = wmma16(a, bfrag(frags, FRAG_WB1 + t, lane), c);
    float b1 = bb1[16 * t + n];
#pragma unroll
    for (int r = 0; r < 8; ++r)
      hm[w][r + 8 * g][16 * t + n] = (_Float16)silu(c[r] + b1);
  }
  WAIT_DS();
  AF a2[2];
#pragma unroll
  for (int kc = 0; kc < 2; ++kc) {
    a2[kc].q[0] = *(const uint4*)&hm[w][n][32 * kc + 8 * g];
    a2[kc].q[1] = *(const uint4*)&hm[w][n][32 * kc + 16 + 8 * g];
  }
#pragma unroll
  for (int t = 0; t < 4; ++t) {
    v8f c = {};
#pragma unroll
    for (int kc = 0; kc < 2; ++kc)
      c = wmma16(a2[kc], bfrag(frags, FRAG_WB2 + kc * 4 + t, lane), c);
    float b2 = bb2[16 * t + n];
#pragma unroll
    for (int r = 0; r < 8; ++r) {
      float v = c[r] + b2;
      size_t o = (size_t)(base + r + 8 * g) * 64 + 16 * t + n;
      ea_out[o] = v;
      if (ea16) ea16[o] = (_Float16)v;
    }
  }
}

// ---------------------------------------------------------------------------
// Edge messages: e=[fi|fj|ea] (K=192) -> phi_s, phi_v (shared A chunks),
// msg_s = phi_s(e)*fi scattered into agg_s; msg_v = phi_v(e)*edge_vec into v0.
// ---------------------------------------------------------------------------
template <bool EA16>
__global__ void __launch_bounds__(256) k_edge_msg(
    const int* edge_index, const float* edge_vec, const _Float16* f16f,
    const float* ea, const _Float16* ea16,
    const float* bs1, const float* bs2,
    const float* bv1, const float* bv2, const _Float16* frags,
    float* aggs, float* v0) {
  int wave = (blockIdx.x * blockDim.x + threadIdx.x) >> 5;
  int lane = threadIdx.x & 31;
  int w = threadIdx.x >> 5;
  int base = wave * 16;
  int g = lane >> 4, n = lane & 15;
  const int* ii = edge_index;
  const int* jj = edge_index + NE;

  int myI = ii[base + n];
  int myJ = jj[base + n];

  // Layer 1 for both MLPs, A chunks shared (6 chunks of K=32).
  v8f as[4] = {{}, {}, {}, {}};
  v8f av[4] = {{}, {}, {}, {}};
#pragma unroll
  for (int kc = 0; kc < 6; ++kc) {
    AF a;
    if (kc < 2)      a_f16(a, f16f + (size_t)myI * 64, 32 * kc, g);
    else if (kc < 4) a_f16(a, f16f + (size_t)myJ * 64, 32 * (kc - 2), g);
    else if (EA16)   a_f16(a, ea16 + (size_t)(base + n) * 64, 32 * (kc - 4), g);
    else             a_f32(a, ea + (size_t)(base + n) * 64, 32 * (kc - 4), g);
#pragma unroll
    for (int t = 0; t < 4; ++t) {
      as[t] = wmma16(a, bfrag(frags, FRAG_WS1 + kc * 4 + t, lane), as[t]);
      av[t] = wmma16(a, bfrag(frags, FRAG_WV1 + kc * 4 + t, lane), av[t]);
    }
  }

  __shared__ __align__(16) _Float16 hm[8][16][64];

  // phi_s layer 2 via LDS transpose bounce.
#pragma unroll
  for (int t = 0; t < 4; ++t) {
    float b1 = bs1[16 * t + n];
#pragma unroll
    for (int r = 0; r < 8; ++r)
      hm[w][r + 8 * g][16 * t + n] = (_Float16)silu(as[t][r] + b1);
  }
  WAIT_DS();
  AF a2[2];
#pragma unroll
  for (int kc = 0; kc < 2; ++kc) {
    a2[kc].q[0] = *(const uint4*)&hm[w][n][32 * kc + 8 * g];
    a2[kc].q[1] = *(const uint4*)&hm[w][n][32 * kc + 16 + 8 * g];
  }
  v8f outs[4];
#pragma unroll
  for (int t = 0; t < 4; ++t) {
    v8f c = {};
#pragma unroll
    for (int kc = 0; kc < 2; ++kc)
      c = wmma16(a2[kc], bfrag(frags, FRAG_WS2 + kc * 4 + t, lane), c);
    outs[t] = c;
  }

  // phi_v layer 2 (reuse LDS region; same-wave LDS ops are in-order).
#pragma unroll
  for (int t = 0; t < 4; ++t) {
    float b1 = bv1[16 * t + n];
#pragma unroll
    for (int r = 0; r < 8; ++r)
      hm[w][r + 8 * g][16 * t + n] = (_Float16)silu(av[t][r] + b1);
  }
  WAIT_DS();
  AF a3[2];
#pragma unroll
  for (int kc = 0; kc < 2; ++kc) {
    a3[kc].q[0] = *(const uint4*)&hm[w][n][32 * kc + 8 * g];
    a3[kc].q[1] = *(const uint4*)&hm[w][n][32 * kc + 16 + 8 * g];
  }
  v8f outv[4];
#pragma unroll
  for (int t = 0; t < 4; ++t) {
    v8f c = {};
#pragma unroll
    for (int kc = 0; kc < 2; ++kc)
      c = wmma16(a3[kc], bfrag(frags, FRAG_WV2 + kc * 4 + t, lane), c);
    outv[t] = c;
  }

  // Epilogue: bias, *fi / *edge_vec, atomic segment-sum (L2 resident).
  int rowI[8], rowJ[8];
  float ev[8][3];
#pragma unroll
  for (int r = 0; r < 8; ++r) {
    int eidx = base + r + 8 * g;
    rowI[r] = ii[eidx];
    rowJ[r] = jj[eidx];
    ev[r][0] = edge_vec[(size_t)eidx * 3 + 0];
    ev[r][1] = edge_vec[(size_t)eidx * 3 + 1];
    ev[r][2] = edge_vec[(size_t)eidx * 3 + 2];
  }
#pragma unroll
  for (int t = 0; t < 4; ++t) {
    int col = 16 * t + n;
    float cbs = bs2[col];
    float cbv = bv2[col];
#pragma unroll
    for (int r = 0; r < 8; ++r) {
      float fiv = (float)f16f[(size_t)rowI[r] * 64 + col];
      float ms = (outs[t][r] + cbs) * fiv;
      unsafeAtomicAdd(&aggs[(size_t)rowJ[r] * 64 + col], ms);
      float mv = outv[t][r] + cbv;
      size_t vb = ((size_t)rowJ[r] * 64 + col) * 3;
      unsafeAtomicAdd(&v0[vb + 0], mv * ev[r][0]);
      unsafeAtomicAdd(&v0[vb + 1], mv * ev[r][1]);
      unsafeAtomicAdd(&v0[vb + 2], mv * ev[r][2]);
    }
  }
}

// ---------------------------------------------------------------------------
// h0 = MLP(concat(f, agg_s); Wh1,bh1,Wh2,bh2).  16 nodes / wave.
// ---------------------------------------------------------------------------
__global__ void __launch_bounds__(256) k_node_update(
    const _Float16* f16f, const float* aggs,
    const float* bh1, const float* bh2,
    const _Float16* frags, float* h0) {
  int wave = (blockIdx.x * blockDim.x + threadIdx.x) >> 5;
  int base = wave * 16;
  if (base >= NN) return;                 // wave-uniform guard (NN = 3125*16)
  int lane = threadIdx.x & 31;
  int w = threadIdx.x >> 5;
  int g = lane >> 4, n = lane & 15;

  v8f acc[4] = {{}, {}, {}, {}};
#pragma unroll
  for (int kc = 0; kc < 4; ++kc) {
    AF a;
    if (kc < 2) a_f16(a, f16f + (size_t)(base + n) * 64, 32 * kc, g);
    else        a_f32(a, aggs + (size_t)(base + n) * 64, 32 * (kc - 2), g);
#pragma unroll
    for (int t = 0; t < 4; ++t)
      acc[t] = wmma16(a, bfrag(frags, FRAG_WH1 + kc * 4 + t, lane), acc[t]);
  }
  __shared__ __align__(16) _Float16 hm[8][16][64];
#pragma unroll
  for (int t = 0; t < 4; ++t) {
    float b1 = bh1[16 * t + n];
#pragma unroll
    for (int r = 0; r < 8; ++r)
      hm[w][r + 8 * g][16 * t + n] = (_Float16)silu(acc[t][r] + b1);
  }
  WAIT_DS();
  AF a2[2];
#pragma unroll
  for (int kc = 0; kc < 2; ++kc) {
    a2[kc].q[0] = *(const uint4*)&hm[w][n][32 * kc + 8 * g];
    a2[kc].q[1] = *(const uint4*)&hm[w][n][32 * kc + 16 + 8 * g];
  }
#pragma unroll
  for (int t = 0; t < 4; ++t) {
    v8f c = {};
#pragma unroll
    for (int kc = 0; kc < 2; ++kc)
      c = wmma16(a2[kc], bfrag(frags, FRAG_WH2 + kc * 4 + t, lane), c);
    float b2 = bh2[16 * t + n];
#pragma unroll
    for (int r = 0; r < 8; ++r)
      h0[(size_t)(base + r + 8 * g) * 64 + 16 * t + n] = c[r] + b2;
  }
}

// ---------------------------------------------------------------------------
extern "C" void kernel_launch(void* const* d_in, const int* in_sizes, int n_in,
                              void* d_out, int out_size, void* d_ws, size_t ws_size,
                              hipStream_t stream) {
  const float* species    = (const float*)d_in[0];
  const int*   edge_index = (const int*)d_in[1];
  const float* edge_attr  = (const float*)d_in[2];
  const float* edge_vec   = (const float*)d_in[3];
  const float* Wa  = (const float*)d_in[4];
  const float* ba  = (const float*)d_in[5];
  const float* Wb1 = (const float*)d_in[6];
  const float* bb1 = (const float*)d_in[7];
  const float* Wb2 = (const float*)d_in[8];
  const float* bb2 = (const float*)d_in[9];
  const float* Ws1 = (const float*)d_in[10];
  const float* bs1 = (const float*)d_in[11];
  const float* Ws2 = (const float*)d_in[12];
  const float* bs2 = (const float*)d_in[13];
  const float* Wh1 = (const float*)d_in[14];
  const float* bh1 = (const float*)d_in[15];
  const float* Wh2 = (const float*)d_in[16];
  const float* bh2 = (const float*)d_in[17];
  const float* Wv1 = (const float*)d_in[18];
  const float* bv1 = (const float*)d_in[19];
  const float* Wv2 = (const float*)d_in[20];
  const float* bv2 = (const float*)d_in[21];

  char* ws = (char*)d_ws;
  _Float16* frags = (_Float16*)(ws + WS_FRAGS);
  _Float16* f16f  = (_Float16*)(ws + WS_F16F);
  float*    aggs  = (float*)(ws + WS_AGGS);
  float*    out   = (float*)d_out;

  bool useEa16 = (ws_size >= WS_NEED_EA16);
  _Float16* ea16 = useEa16 ? (_Float16*)(ws + WS_EA16) : (_Float16*)nullptr;

  k_prep<<<N_FRAGS, 32, 0, stream>>>(Wb1, Wb2, Ws1, Ws2, Wh1, Wh2, Wv1, Wv2, frags);
  k_zero<<<4096, 256, 0, stream>>>(aggs, out + OUT_V0);
  k_node_embed<<<(NN * 64 + 255) / 256, 256, 0, stream>>>(species, Wa, ba, f16f);
  k_edge_embed<<<NE / 128, 256, 0, stream>>>(edge_attr, bb1, bb2, frags,
                                             out + OUT_EA, ea16);
  if (useEa16)
    k_edge_msg<true><<<NE / 128, 256, 0, stream>>>(edge_index, edge_vec, f16f,
                                                   out + OUT_EA, ea16,
                                                   bs1, bs2, bv1, bv2, frags,
                                                   aggs, out + OUT_V0);
  else
    k_edge_msg<false><<<NE / 128, 256, 0, stream>>>(edge_index, edge_vec, f16f,
                                                    out + OUT_EA, ea16,
                                                    bs1, bs2, bv1, bv2, frags,
                                                    aggs, out + OUT_V0);
  k_node_update<<<(NN + 127) / 128, 256, 0, stream>>>(f16f, aggs, bh1, bh2, frags,
                                                      out + OUT_H0);
}
